// TriangleAttention_88115549045152
// MI455X (gfx1250) — matrix operand order repaired and checked by reading, not compiled, not run on hardware
//
#include <hip/hip_runtime.h>
#include <hip/hip_bf16.h>

typedef __attribute__((ext_vector_type(16))) _Float16 v16h;
typedef __attribute__((ext_vector_type(8)))  _Float16 v8h;
typedef __attribute__((ext_vector_type(8)))  float    v8f;

#define WMMA_F16(a, b, c) \
  __builtin_amdgcn_wmma_f32_16x16x32_f16(false, (a), false, (b), (short)0, (c), false, false)

#define NQ   256        // rows in i and j (N)
#define DD   128        // model dim
#define NH   4          // heads
#define DH   32         // head dim
#define NEG_INF (-3.4028235e38f)

// A-fragment: lane = M = L%16, half = L/16; elements 0..15 -> K = 16*half + e,
// i.e. K runs 16*half .. 16*half+15  => two contiguous 8-half chunks.
__device__ __forceinline__ v16h cat16(v8h a, v8h b) {
  return __builtin_shufflevector(a, b, 0, 1, 2, 3, 4, 5, 6, 7,
                                       8, 9, 10, 11, 12, 13, 14, 15);
}

// ---------------------------------------------------------------------------
// Kernel 0: convert + swizzle weights fp32 -> f16 B-fragment order.
// WB[(((kc*8+ct)*32 + L)*16 + e] = W[(kc*32 + e + 16*(L/16))*128 + ct*16 + L%16]
// so a lane's whole B fragment is one contiguous 32-byte chunk.
// ---------------------------------------------------------------------------
__global__ __launch_bounds__(256) void wconv_k(
    const float* __restrict__ Wq, const float* __restrict__ Wk,
    const float* __restrict__ Wv, const float* __restrict__ Wo,
    const float* __restrict__ Wg, _Float16* __restrict__ out) {
  int idx = blockIdx.x * 256 + threadIdx.x;       // 5 * 16384 total
  int m = idx >> 14;
  int r = idx & 16383;
  int e    = r & 15;
  int L    = (r >> 4) & 31;
  int tile = r >> 9;           // 0..31 = kc*8 + ct
  int kc = tile >> 3, ct = tile & 7;
  const float* src = (m == 0) ? Wq : (m == 1) ? Wk : (m == 2) ? Wv : (m == 3) ? Wo : Wg;
  out[idx] = (_Float16)src[(kc * 32 + e + 16 * (L >> 4)) * DD + ct * 16 + (L & 15)];
}

// ---------------------------------------------------------------------------
// Kernel 1: bias[j,h] = mean_i(LN(z)[i,j,:]) @ Wb      (tiny, one block per j)
// ---------------------------------------------------------------------------
__global__ __launch_bounds__(256) void bias_col_k(
    const float* __restrict__ z, const float* __restrict__ gamma,
    const float* __restrict__ beta, const float* __restrict__ Wb,
    float* __restrict__ biasOut) {
  int j = blockIdx.x;
  int t = threadIdx.x;            // thread t handles i = t
  __shared__ float acc[DD];
  if (t < DD) acc[t] = 0.f;
  __syncthreads();

  const float* row = z + ((size_t)t * NQ + j) * DD;
  float s = 0.f;
  for (int c = 0; c < DD; ++c) s += row[c];
  float mean = s * (1.f / DD);
  float s2 = 0.f;
  for (int c = 0; c < DD; ++c) { float d = row[c] - mean; s2 += d * d; }
  float rstd = rsqrtf(s2 * (1.f / DD) + 1e-5f);
  for (int c = 0; c < DD; ++c) {
    float nv = (row[c] - mean) * rstd * gamma[c] + beta[c];
    atomicAdd(&acc[c], nv);
  }
  __syncthreads();
  if (t < NH) {
    float d = 0.f;
    for (int c = 0; c < DD; ++c) d += (acc[c] * (1.f / NQ)) * Wb[c * NH + t];
    biasOut[j * NH + t] = d;
  }
}

// ---------------------------------------------------------------------------
// Kernel 2: fused LayerNorm + q/k/v projections (WMMA), 16 rows per block
// ---------------------------------------------------------------------------
__global__ __launch_bounds__(256) void ln_qkv_k(
    const float* __restrict__ z, const float* __restrict__ gamma,
    const float* __restrict__ beta,
    const float* __restrict__ bq, const float* __restrict__ bk,
    const float* __restrict__ bv,
    const _Float16* __restrict__ WqH, const _Float16* __restrict__ WkH,
    const _Float16* __restrict__ WvH,
    _Float16* __restrict__ qo, _Float16* __restrict__ ko,
    _Float16* __restrict__ vo) {
  __shared__ float zf[16][DD];
  __shared__ _Float16 zn[16][136];   // 68-dword row stride: conflict-free, 16B-aligned
  int m0 = blockIdx.x * 16;
  int t = threadIdx.x;

  const float4* zsrc = (const float4*)(z + (size_t)m0 * DD);
  float4* zdst = (float4*)(&zf[0][0]);
  for (int it = t; it < 512; it += 256) zdst[it] = zsrc[it];
  __syncthreads();

  // LayerNorm: 16 threads per row, xor-shuffle reduce within 16-lane halves
  int row = t >> 4, sub = t & 15;
  float s = 0.f;
#pragma unroll
  for (int e = 0; e < 8; ++e) s += zf[row][sub * 8 + e];
#pragma unroll
  for (int mm = 1; mm < 16; mm <<= 1) s += __shfl_xor(s, mm, 32);
  float mean = s * (1.f / DD);
  float s2 = 0.f;
#pragma unroll
  for (int e = 0; e < 8; ++e) { float d = zf[row][sub * 8 + e] - mean; s2 += d * d; }
#pragma unroll
  for (int mm = 1; mm < 16; mm <<= 1) s2 += __shfl_xor(s2, mm, 32);
  float rstd = rsqrtf(s2 * (1.f / DD) + 1e-5f);
#pragma unroll
  for (int e = 0; e < 8; ++e) {
    int c = sub * 8 + e;
    zn[row][c] = (_Float16)((zf[row][c] - mean) * rstd * gamma[c] + beta[c]);
  }
  __syncthreads();

  int lane = t & 31, w = t >> 5, half = lane >> 4, lm = lane & 15;
#pragma unroll
  for (int job = w; job < 24; job += 8) {
    int mat = job >> 3, ct = job & 7;
    const v16h* WB = (const v16h*)((mat == 0) ? WqH : (mat == 1) ? WkH : WvH);
    const float* bvec = (mat == 0) ? bq : (mat == 1) ? bk : bv;
    _Float16* outp    = (mat == 0) ? qo : (mat == 1) ? ko : vo;
    v8f c = {};
#pragma unroll
    for (int kc = 0; kc < 4; ++kc) {
      v8h a0 = *(const v8h*)&zn[lm][kc * 32 + 16 * half];
      v8h a1 = *(const v8h*)&zn[lm][kc * 32 + 16 * half + 8];
      v16h b = WB[(kc * 8 + ct) * 32 + lane];     // 32B contiguous per lane
      c = WMMA_F16(cat16(a0, a1), b, c);
    }
#pragma unroll
    for (int r = 0; r < 8; ++r) {
      int col = ct * 16 + lm;
      outp[(size_t)(m0 + r + 8 * half) * DD + col] = (_Float16)(c[r] + bvec[col]);
    }
  }
}

// ---------------------------------------------------------------------------
// Kernel 3: attention, one block per (i, h), 16 waves, flash-style softmax
// ---------------------------------------------------------------------------
__global__ __launch_bounds__(512) void attn_k(
    const _Float16* __restrict__ q, const _Float16* __restrict__ k,
    const _Float16* __restrict__ v, const float* __restrict__ bias,
    const unsigned char* __restrict__ mask, _Float16* __restrict__ aout) {
  int i = blockIdx.x, h = blockIdx.y;
  __shared__ _Float16 Kh[NQ][40];          // 20 KB: row-major, B-frag rows contiguous
  __shared__ _Float16 VhT[DH][264];        // 16.5 KB: transposed V -> contiguous B-frags
  __shared__ _Float16 Ps[16 * 16 * 40];    // 20 KB: per-wave 16x32 P scratch

  int t = threadIdx.x;
  {
    int row = t >> 1, off = (t & 1) * 16;
    const float4* ks = (const float4*)(k + ((size_t)(i * NQ + row)) * DD + h * DH + off);
    float4* kd = (float4*)(&Kh[row][off]);
    kd[0] = ks[0]; kd[1] = ks[1];
    const v8h* vs = (const v8h*)(v + ((size_t)(i * NQ + row)) * DD + h * DH + off);
    v8h va = vs[0], vb = vs[1];
#pragma unroll
    for (int d = 0; d < 8; ++d) VhT[off + d][row] = va[d];
#pragma unroll
    for (int d = 0; d < 8; ++d) VhT[off + 8 + d][row] = vb[d];
  }
  __syncthreads();

  int lane = t & 31, w = t >> 5, half = lane >> 4, lm = lane & 15;
  int q0 = w * 16;                          // this wave's query stripe
  const float SCL = 0.17677669529663687f;   // 1/sqrt(32)

  // Q A-fragment: two contiguous 16B chunks per lane
  const _Float16* qb = q + ((size_t)(i * NQ) + q0 + lm) * DD + h * DH;
  v16h aQ = cat16(*(const v8h*)(qb + 16 * half), *(const v8h*)(qb + 16 * half + 8));

  float biasr[8];
#pragma unroll
  for (int r = 0; r < 8; ++r) biasr[r] = bias[(q0 + r + 8 * half) * NH + h];

  float m_[8], l_[8];
  v8f o0 = {}, o1 = {};
#pragma unroll
  for (int r = 0; r < 8; ++r) { m_[r] = NEG_INF; l_[r] = 0.f; }

  for (int kt2 = 0; kt2 < 8; ++kt2) {       // 32 keys per chunk
    v8f sc[2];
    float rmax[8];
#pragma unroll
    for (int r = 0; r < 8; ++r) rmax[r] = NEG_INF;

#pragma unroll
    for (int sub = 0; sub < 2; ++sub) {
      int kt = kt2 * 2 + sub;
      const _Float16* kr = &Kh[kt * 16 + lm][16 * half];   // 16 contiguous halves
      v16h bK = cat16(*(const v8h*)kr, *(const v8h*)(kr + 8));
      v8f c = {};
      c = WMMA_F16(aQ, bK, c);
      bool mok = mask[kt * 16 + lm] != 0;   // key column = lane's N index
#pragma unroll
      for (int r = 0; r < 8; ++r) {
        float sv = c[r] * SCL + biasr[r];
        c[r] = mok ? sv : NEG_INF;
      }
      sc[sub] = c;
#pragma unroll
      for (int r = 0; r < 8; ++r) {
        float vx = c[r];
#pragma unroll
        for (int mm = 1; mm < 16; mm <<= 1) vx = fmaxf(vx, __shfl_xor(vx, mm, 32));
        rmax[r] = fmaxf(rmax[r], vx);
      }
    }

    // online-softmax rescale
#pragma unroll
    for (int r = 0; r < 8; ++r) {
      float mnew = fmaxf(m_[r], rmax[r]);
      float f = __expf(m_[r] - mnew);
      m_[r] = mnew;
      l_[r] *= f;
      o0[r] *= f;
      o1[r] *= f;
    }

    __builtin_amdgcn_wave_barrier();
#pragma unroll
    for (int sub = 0; sub < 2; ++sub) {
#pragma unroll
      for (int r = 0; r < 8; ++r) {
        float p = __expf(sc[sub][r] - m_[r]);
        float ps = p;
#pragma unroll
        for (int mm = 1; mm < 16; mm <<= 1) ps += __shfl_xor(ps, mm, 32);
        l_[r] += ps;
        Ps[w * 640 + (r + 8 * half) * 40 + sub * 16 + lm] = (_Float16)p;
      }
    }
    __builtin_amdgcn_wave_barrier();

    // read back P in A-layout (two contiguous chunks), P @ V with contiguous B
    const _Float16* pb = &Ps[w * 640 + lm * 40];
    v16h aP = cat16(*(const v8h*)(pb + 16 * half), *(const v8h*)(pb + 16 * half + 8));
    const _Float16* v0 = &VhT[lm][kt2 * 32 + 16 * half];
    const _Float16* v1 = &VhT[16 + lm][kt2 * 32 + 16 * half];
    v16h bV0 = cat16(*(const v8h*)v0, *(const v8h*)(v0 + 8));
    v16h bV1 = cat16(*(const v8h*)v1, *(const v8h*)(v1 + 8));
    o0 = WMMA_F16(aP, bV0, o0);
    o1 = WMMA_F16(aP, bV1, o1);
    __builtin_amdgcn_wave_barrier();
  }

  // normalize and store [i, j, h*32 + dh] as f16
#pragma unroll
  for (int r = 0; r < 8; ++r) {
    float inv = 1.f / l_[r];
    size_t base = ((size_t)(i * NQ) + q0 + r + 8 * half) * DD + h * DH;
    aout[base + lm]      = (_Float16)(o0[r] * inv);
    aout[base + 16 + lm] = (_Float16)(o1[r] * inv);
  }
}

// ---------------------------------------------------------------------------
// Kernel 4: out = sigmoid(z@Wg + bg) * (attn@Wo + bo), fp32 store
// ---------------------------------------------------------------------------
__global__ __launch_bounds__(256) void gate_out_k(
    const float* __restrict__ z, const _Float16* __restrict__ af,
    const _Float16* __restrict__ WoH, const _Float16* __restrict__ WgH,
    const float* __restrict__ bg, const float* __restrict__ bo,
    float* __restrict__ out) {
  __shared__ _Float16 zg[16][136];
  __shared__ _Float16 ao[16][136];
  int m0 = blockIdx.x * 16;
  int t = threadIdx.x;

  const float4* zs = (const float4*)(z + (size_t)m0 * DD);
  for (int it = t; it < 512; it += 256) {
    float4 vv = zs[it];
    int r = it >> 5, c = (it & 31) * 4;
    zg[r][c] = (_Float16)vv.x; zg[r][c + 1] = (_Float16)vv.y;
    zg[r][c + 2] = (_Float16)vv.z; zg[r][c + 3] = (_Float16)vv.w;
  }
  const float4* as_ = (const float4*)(af + (size_t)m0 * DD);
  for (int it = t; it < 256; it += 256) {
    int r = it >> 4, c = (it & 15) * 8;
    *(float4*)&ao[r][c] = as_[it];
  }
  __syncthreads();

  int lane = t & 31, w = t >> 5, half = lane >> 4, lm = lane & 15;
  int ct = w;
  const v16h* WGB = (const v16h*)WgH;
  const v16h* WOB = (const v16h*)WoH;
  v8f cg = {}, co = {};
#pragma unroll
  for (int kc = 0; kc < 4; ++kc) {
    v16h ag  = cat16(*(const v8h*)&zg[lm][kc * 32 + 16 * half],
                     *(const v8h*)&zg[lm][kc * 32 + 16 * half + 8]);
    v16h aoF = cat16(*(const v8h*)&ao[lm][kc * 32 + 16 * half],
                     *(const v8h*)&ao[lm][kc * 32 + 16 * half + 8]);
    v16h bG = WGB[(kc * 8 + ct) * 32 + lane];
    v16h bO = WOB[(kc * 8 + ct) * 32 + lane];
    cg = WMMA_F16(ag, bG, cg);
    co = WMMA_F16(aoF, bO, co);
  }
#pragma unroll
  for (int r = 0; r < 8; ++r) {
    int col = ct * 16 + lm;
    int mrow = m0 + r + 8 * half;
    float g = 1.f / (1.f + __expf(-(cg[r] + bg[col])));
    out[(size_t)mrow * DD + col] = g * (co[r] + bo[col]);
  }
}

// ---------------------------------------------------------------------------
extern "C" void kernel_launch(void* const* d_in, const int* in_sizes, int n_in,
                              void* d_out, int out_size, void* d_ws, size_t ws_size,
                              hipStream_t stream) {
  const float* z        = (const float*)d_in[0];
  const unsigned char* mask = (const unsigned char*)d_in[1];
  const float* ln_gamma = (const float*)d_in[2];
  const float* ln_beta  = (const float*)d_in[3];
  const float* Wq = (const float*)d_in[4];  const float* bq = (const float*)d_in[5];
  const float* Wk = (const float*)d_in[6];  const float* bk = (const float*)d_in[7];
  const float* Wv = (const float*)d_in[8];  const float* bv = (const float*)d_in[9];
  const float* Wb = (const float*)d_in[10];
  const float* Wg = (const float*)d_in[11]; const float* bg = (const float*)d_in[12];
  const float* Wo = (const float*)d_in[13]; const float* bo = (const float*)d_in[14];
  float* out = (float*)d_out;

  const size_t SZ = (size_t)NQ * NQ * DD;           // 8,388,608 elements
  _Float16* qf = (_Float16*)d_ws;
  _Float16* kf = qf + SZ;
  _Float16* vf = kf + SZ;
  _Float16* af = vf + SZ;
  _Float16* wf = af + SZ;                           // 5 * 16384 f16 weights (swizzled)
  float* biasB = (float*)(wf + 5 * 16384);          // 256*4 fp32

  _Float16* WqH = wf + 0 * 16384;
  _Float16* WkH = wf + 1 * 16384;
  _Float16* WvH = wf + 2 * 16384;
  _Float16* WoH = wf + 3 * 16384;
  _Float16* WgH = wf + 4 * 16384;

  wconv_k<<<320, 256, 0, stream>>>(Wq, Wk, Wv, Wo, Wg, wf);
  bias_col_k<<<256, 256, 0, stream>>>(z, ln_gamma, ln_beta, Wb, biasB);
  ln_qkv_k<<<NQ * NQ / 16, 256, 0, stream>>>(z, ln_gamma, ln_beta, bq, bk, bv,
                                             WqH, WkH, WvH, qf, kf, vf);
  attn_k<<<dim3(NQ, NH), 512, 0, stream>>>(qf, kf, vf, biasB, mask, af);
  gate_out_k<<<NQ * NQ / 16, 256, 0, stream>>>(z, af, WoH, WgH, bg, bo, out);
}
